// DotProductAttention_18794776888015
// MI455X (gfx1250) — matrix operand order repaired and checked by reading, not compiled, not run on hardware
//
#include <hip/hip_runtime.h>
#include <hip/hip_bf16.h>

#define BHN   64
#define SEQ   1024
#define HD    64
#define ALPHA 0.99f

typedef __attribute__((ext_vector_type(16))) _Float16 v16h;
typedef __attribute__((ext_vector_type(8)))  _Float16 v8h;
typedef __attribute__((ext_vector_type(8)))  float    v8f;

union Frag16 { v16h v; v8h h[2]; };

#define QP 72   // pitch (in halfs) for q_s / k_s rows of 64 dims (144B, mult of 16)
#define VP 40   // pitch (in halfs) for v_s / p_s rows of 32 keys (80B, mult of 16)

__launch_bounds__(128, 2)
__global__ void DotProductAttention_18794776888015_kernel(
    const float* __restrict__ q, const float* __restrict__ k,
    const float* __restrict__ v, const float* __restrict__ mask,
    float* __restrict__ out)
{
  __shared__ __align__(16) _Float16 q_s[64 * QP];     // [qrow][d]
  __shared__ __align__(16) _Float16 k_s[32 * QP];     // [key][d]
  __shared__ __align__(16) _Float16 v_s[64 * VP];     // transposed: [d][key]
  __shared__ __align__(16) _Float16 p_s[4][16 * VP];  // per-wave probs [qrow][key]

  const int tid   = threadIdx.x;
  const int lane  = tid & 31;
  const int wv    = tid >> 5;
  const int b     = blockIdx.x >> 4;          // head index
  const int qbase = (blockIdx.x & 15) * 64;   // 64 query rows per block

  const int half = lane >> 4;                 // which half-wave
  const int ln   = lane & 15;

  // ---- stage Q tile (64x64), fp32 -> fp16 ----
  {
    const float* qg = q + ((size_t)b * SEQ + qbase) * HD;
    for (int e = tid; e < 64 * 64; e += 128) {
      int r = e >> 6, d = e & 63;
      q_s[r * QP + d] = (_Float16)qg[r * HD + d];
    }
  }
  __syncthreads();

  // ---- per-wave Q A-fragments: 16 rows x 32 K, two D-chunks ----
  Frag16 aq[2];
  {
    const int row = wv * 16 + ln;
    const int ka  = half * 8;                 // A layout: K in {ka..ka+7, 16+ka..}
    for (int ch = 0; ch < 2; ++ch) {
      const int base = ch * 32;
      aq[ch].h[0] = *(const v8h*)&q_s[row * QP + base + ka];
      aq[ch].h[1] = *(const v8h*)&q_s[row * QP + base + 16 + ka];
    }
  }

  v8f acc[4];
  float mrun[8], lrun[8];
  for (int t = 0; t < 4; ++t) acc[t] = (v8f)0.0f;
  for (int c = 0; c < 8; ++c) { mrun[c] = -1e30f; lrun[c] = 0.0f; }

  // decay(k) = log(ALPHA^(S-1-k) + 1e-10) ~= (S-1-k)*ln(ALPHA)
  // (ALPHA^n >= 3.4e-5 >> 1e-10, correction <= 3e-6: below f16-matmul noise)
  const float ln_alpha   = __logf(ALPHA);
  const float decay_base = (float)(SEQ - 1) * ln_alpha;

  for (int kt = 0; kt < SEQ; kt += 32) {
    __syncthreads();
    // ---- stage K (row-major) and V (transposed) 32-key tiles ----
    {
      const float* kg = k + ((size_t)b * SEQ + kt) * HD;
      const float* vg = v + ((size_t)b * SEQ + kt) * HD;
      for (int e = tid; e < 32 * 64; e += 128) {
        int r = e >> 6, d = e & 63;
        k_s[r * QP + d] = (_Float16)kg[r * HD + d];
        v_s[d * VP + r] = (_Float16)vg[r * HD + d];
      }
    }
    __syncthreads();

    // ---- scores: two 16-key sub-tiles, accumulate over D in 2 chunks ----
    v8f sc[2];
    #pragma unroll
    for (int sub = 0; sub < 2; ++sub) {
      const int key = sub * 16 + ln;          // B column = key
      const int kb  = half * 8;               // B layout: K in {kb..kb+7, 16+kb..}
      v8f c8 = (v8f)0.0f;
      #pragma unroll
      for (int ch = 0; ch < 2; ++ch) {
        Frag16 bk;
        const int base = ch * 32 + kb;
        bk.h[0] = *(const v8h*)&k_s[key * QP + base];
        bk.h[1] = *(const v8h*)&k_s[key * QP + base + 16];
        c8 = __builtin_amdgcn_wmma_f32_16x16x32_f16(false, aq[ch].v, false, bk.v,
                                                    (short)0, c8, false, false);
      }
      // add mask + log-decay bias (single FMA, no exp/log in the loop)
      const int key_g   = kt + sub * 16 + ln;
      const float decay = fmaf(-(float)key_g, ln_alpha, decay_base);
      const float* mrow = mask + (size_t)b * SEQ * SEQ
                        + (size_t)(qbase + wv * 16 + half * 8) * SEQ + key_g;
      #pragma unroll
      for (int c = 0; c < 8; ++c)
        c8[c] += mrow[(size_t)c * SEQ] + decay;
      sc[sub] = c8;
    }

    // ---- online softmax update (row = component c + 8*half) ----
    #pragma unroll
    for (int c = 0; c < 8; ++c) {
      float tmax = fmaxf(sc[0][c], sc[1][c]);
      #pragma unroll
      for (int off = 8; off > 0; off >>= 1)
        tmax = fmaxf(tmax, __shfl_xor(tmax, off, 16));
      const float mn    = fmaxf(mrun[c], tmax);
      const float scale = __expf(mrun[c] - mn);
      const float p0 = __expf(sc[0][c] - mn);
      const float p1 = __expf(sc[1][c] - mn);
      float ts = p0 + p1;
      #pragma unroll
      for (int off = 8; off > 0; off >>= 1)
        ts += __shfl_xor(ts, off, 16);
      lrun[c] = lrun[c] * scale + ts;
      mrun[c] = mn;
      #pragma unroll
      for (int t = 0; t < 4; ++t) acc[t][c] *= scale;
      // spill probs to LDS (f16) for the PV matmul A-fragment
      p_s[wv][(c + half * 8) * VP + ln]      = (_Float16)p0;
      p_s[wv][(c + half * 8) * VP + 16 + ln] = (_Float16)p1;
    }

    // ---- PV: acc += P(16x32) x V(32x64-tile), 4 output column tiles ----
    Frag16 ap;
    {
      const int ka = half * 8;
      ap.h[0] = *(const v8h*)&p_s[wv][ln * VP + ka];
      ap.h[1] = *(const v8h*)&p_s[wv][ln * VP + 16 + ka];
    }
    #pragma unroll
    for (int t = 0; t < 4; ++t) {
      Frag16 bv;
      const int dcol = t * 16 + ln;
      const int kb   = half * 8;
      bv.h[0] = *(const v8h*)&v_s[dcol * VP + kb];
      bv.h[1] = *(const v8h*)&v_s[dcol * VP + kb + 16];
      acc[t] = __builtin_amdgcn_wmma_f32_16x16x32_f16(false, ap.v, false, bv.v,
                                                      (short)0, acc[t], false, false);
    }
  }

  // ---- epilogue: normalize by row sums and store fp32 ----
  {
    float* og = out + ((size_t)b * SEQ + qbase + wv * 16) * HD;
    #pragma unroll
    for (int c = 0; c < 8; ++c) {
      const float inv = 1.0f / lrun[c];
      const int row   = c + half * 8;
      #pragma unroll
      for (int t = 0; t < 4; ++t)
        og[(size_t)row * HD + t * 16 + ln] = acc[t][c] * inv;
    }
  }
}

extern "C" void kernel_launch(void* const* d_in, const int* in_sizes, int n_in,
                              void* d_out, int out_size, void* d_ws, size_t ws_size,
                              hipStream_t stream) {
  const float* q    = (const float*)d_in[0];
  const float* k    = (const float*)d_in[1];
  const float* v    = (const float*)d_in[2];
  const float* mask = (const float*)d_in[3];
  float* out        = (float*)d_out;

  // grid: BH heads x (SEQ/64) query blocks; 128 threads = 4 waves (wave32)
  dim3 grid(BHN * (SEQ / 64));
  dim3 block(128);
  DotProductAttention_18794776888015_kernel<<<grid, block, 0, stream>>>(q, k, v, mask, out);
}
